// MHSA_40346922778634
// MI455X (gfx1250) — matrix operand order repaired and checked by reading, not compiled, not run on hardware
//
#include <hip/hip_runtime.h>
#include <hip/hip_bf16.h>
#include <stdint.h>

// ---------------------------------------------------------------------------
// MHSA (B=16, C=512, 32x32, 8 heads, d=64) for gfx1250 using bf16 WMMA.
// The reg_qk/reg_v "9th head" is dropped by out[:, :head] -> dead inputs.
// ---------------------------------------------------------------------------

#define C_DIM  512
#define NPIX   1024
#define DH     64
#define NHEAD  8
#define BATCH  16
#define OCH    (3 * C_DIM)   // 1536 fused q/k/v output channels

// LDS layout for the attention kernel (all rows padded to dodge bank conflicts)
#define S_PITCH   1040                 // floats per S row   (1024 + 16)
#define PR_PITCH  1032                 // ushorts per prob row (1024 + 8)
#define V_PITCH   1032                 // ushorts per V row in LDS (1024 + 8)
#define S_BYTES   (32 * S_PITCH * 4)   // 133120
#define V_BYTES   (64 * V_PITCH * 2)   // 132096 (overlays retired S region)
#define PR_OFF    S_BYTES              // probs live after the S/V union
#define SMEM_BYTES (PR_OFF + 32 * PR_PITCH * 2)   // 199168 < 320 KB

typedef __bf16 v16bf __attribute__((ext_vector_type(16)));
typedef float  v8f   __attribute__((ext_vector_type(8)));

struct U2 { uint4 a, b; };

// float -> bf16 (round-to-nearest-even), as raw ushort bits
static __device__ inline unsigned short f2bf(float f) {
    unsigned u = __float_as_uint(f);
    unsigned r = (u + 0x7FFFu + ((u >> 16) & 1u)) >> 16;
    return (unsigned short)r;
}

// A-fragment (16x32 bf16): per-lane elements K = k0..k0+7 and k0+16..k0+23
// caller passes p already offset by (row_base + ks*32 + (lane>=16 ? 8 : 0))
static __device__ inline v16bf ldfragA(const unsigned short* p) {
    U2 t{ *(const uint4*)(p), *(const uint4*)(p + 16) };
    return __builtin_bit_cast(v16bf, t);
}
// B-fragment (32x16 bf16): per-lane elements K = k0..k0+7 and k0+16..k0+23
// caller passes p already offset by (col_base + ks*32 + (lane>=16 ? 8 : 0))
static __device__ inline v16bf ldfragB(const unsigned short* p) {
    U2 t{ *(const uint4*)(p), *(const uint4*)(p + 16) };
    return __builtin_bit_cast(v16bf, t);
}

static __device__ inline v8f bfwmma(v16bf a, v16bf b, v8f c) {
    return __builtin_amdgcn_wmma_f32_16x16x32_bf16(
        /*neg_a=*/false, a, /*neg_b=*/false, b,
        /*c_mod=*/(short)0, c, /*reuse_a=*/false, /*reuse_b=*/false);
}

// ---------------------------------------------------------------------------
// Prep kernels: convert f32 inputs into bf16 workspace layouts
// ---------------------------------------------------------------------------

// x [B][C][N] f32  ->  xT [B][N][C] bf16   (A-side of QKV GEMM)
__global__ void prep_x_kernel(const float* __restrict__ x,
                              unsigned short* __restrict__ xT) {
    size_t idx = (size_t)blockIdx.x * blockDim.x + threadIdx.x; // B*N*C
    int c = (int)(idx & (C_DIM - 1));
    size_t bn = idx >> 9;
    int n = (int)(bn & (NPIX - 1));
    int b = (int)(bn >> 10);
    xT[idx] = f2bf(x[((size_t)b * C_DIM + c) * NPIX + n]);
}

// Wq/Wk/Wv [512][512] f32 -> Wcat [1536][512] bf16; biases -> biascat f32[1536]
__global__ void prep_w_kernel(const float* __restrict__ Wq, const float* __restrict__ Wk,
                              const float* __restrict__ Wv, const float* __restrict__ bq,
                              const float* __restrict__ bk, const float* __restrict__ bv,
                              unsigned short* __restrict__ Wcat, float* __restrict__ biascat) {
    int idx = blockIdx.x * blockDim.x + threadIdx.x; // OCH*C_DIM
    int o = idx >> 9, c = idx & (C_DIM - 1);
    const float* src = (o < 512) ? Wq : (o < 1024 ? Wk : Wv);
    Wcat[idx] = f2bf(src[(o & 511) * C_DIM + c]);
    if (idx < OCH) {
        const float* bs = (idx < 512) ? bq : (idx < 1024 ? bk : bv);
        biascat[idx] = bs[idx & 511];
    }
}

// pos = rel_h + rel_w broadcast -> PT [head][N][d] bf16 (transposed, d contiguous)
__global__ void prep_p_kernel(const float* __restrict__ rel_h,
                              const float* __restrict__ rel_w,
                              unsigned short* __restrict__ PT) {
    int idx = blockIdx.x * blockDim.x + threadIdx.x; // NHEAD*NPIX*DH
    int dd = idx & (DH - 1);
    int n  = (idx >> 6) & (NPIX - 1);
    int h  = idx >> 16;
    int w = n >> 5, hh = n & 31;
    float v = rel_h[(h * DH + dd) * 32 + hh] + rel_w[(h * DH + dd) * 32 + w];
    PT[idx] = f2bf(v);
}

// ---------------------------------------------------------------------------
// Fused QKV projection GEMM:  out[o][n] = sum_c Wcat[o][c] * x[b][c][n] + bias
// One wave per 16x16 output tile (M = n rows, N = o cols), K = 512.
// Epilogue: Q,K stored transposed [b][h][n][d]; V stored [b][h][d][n].
// ---------------------------------------------------------------------------
__global__ __launch_bounds__(256) void qkv_gemm_kernel(
    const unsigned short* __restrict__ xT, const unsigned short* __restrict__ Wcat,
    const float* __restrict__ bias,
    unsigned short* __restrict__ QT, unsigned short* __restrict__ KT,
    unsigned short* __restrict__ Vd) {
    const int wid  = blockIdx.x * 8 + ((int)threadIdx.x >> 5);
    const int lane = (int)threadIdx.x & 31;
    const int l16 = lane & 15, lhalf = lane >> 4;

    const int nt = wid & 63;           // 64 n-tiles
    const int t2 = wid >> 6;
    const int ot = t2 % 96;            // 96 o-tiles
    const int b  = t2 / 96;            // 16 batches

    const unsigned short* Arow = xT   + ((size_t)b * NPIX + nt * 16 + l16) * C_DIM;
    const unsigned short* Brow = Wcat + (size_t)(ot * 16 + l16) * C_DIM;

    v8f acc = { 0.f, 0.f, 0.f, 0.f, 0.f, 0.f, 0.f, 0.f };
    #pragma unroll
    for (int ks = 0; ks < 16; ++ks) {
        v16bf A = ldfragA(Arow + ks * 32 + lhalf * 8);
        v16bf B = ldfragB(Brow + ks * 32 + lhalf * 8);
        acc = bfwmma(A, B, acc);
    }

    const int o  = ot * 16 + l16;      // output channel (per-lane column)
    const float bval = bias[o];
    const int nb = nt * 16 + lhalf * 8; // first output row this lane holds

    if (o < 1024) {  // Q or K -> transposed [b][h][n][d]
        unsigned short* dst = (o < 512) ? QT : KT;
        const int oc = o & 511;
        const int hh = oc >> 6, dd = oc & (DH - 1);
        unsigned short* p = dst + (((size_t)(b * NHEAD + hh) * NPIX) + nb) * DH + dd;
        #pragma unroll
        for (int r = 0; r < 8; ++r)
            p[(size_t)r * DH] = f2bf(acc[r] + bval);
    } else {         // V -> natural [b][h][d][n], 8 consecutive n -> one b128 store
        const int oc = o - 1024;
        const int hh = oc >> 6, dd = oc & (DH - 1);
        unsigned short* p = Vd + ((size_t)(b * NHEAD + hh) * DH + dd) * NPIX + nb;
        unsigned w0 = (unsigned)f2bf(acc[0] + bval) | ((unsigned)f2bf(acc[1] + bval) << 16);
        unsigned w1 = (unsigned)f2bf(acc[2] + bval) | ((unsigned)f2bf(acc[3] + bval) << 16);
        unsigned w2 = (unsigned)f2bf(acc[4] + bval) | ((unsigned)f2bf(acc[5] + bval) << 16);
        unsigned w3 = (unsigned)f2bf(acc[6] + bval) | ((unsigned)f2bf(acc[7] + bval) << 16);
        *(uint4*)p = make_uint4(w0, w1, w2, w3);
    }
}

// ---------------------------------------------------------------------------
// Fused attention: one block per (b, h, 32-row slab).
// Phase 1: S[32][1024] = Q^T K + P^T Q   (kept in LDS, f32, padded pitch)
// Phase 2: row softmax (8-lane-group __shfl_xor reduce), bf16 probs in LDS
// Phase V: async-stage V[64][1024] bf16 into the retired S region
//          (global_load_async_to_lds_b128 + s_wait_asynccnt)
// Phase 3: O[32][64] = probs @ V^T entirely from LDS, + residual, write d_out
// ---------------------------------------------------------------------------
__global__ __launch_bounds__(256) void attn_kernel(
    const unsigned short* __restrict__ QT, const unsigned short* __restrict__ KT,
    const unsigned short* __restrict__ Vd, const unsigned short* __restrict__ PT,
    const float* __restrict__ x, float* __restrict__ out) {
    extern __shared__ char smem[];
    float* S = (float*)smem;                                  // [32][S_PITCH]
    unsigned short* Pr = (unsigned short*)(smem + PR_OFF);    // [32][PR_PITCH]

    const int bh   = (int)blockIdx.x >> 5;   // 0..127 = b*8+h
    const int nblk = (int)blockIdx.x & 31;
    const int b = bh >> 3, h = bh & 7;
    const int n0 = nblk * 32;

    const unsigned short* Qbh = QT + (size_t)bh * NPIX * DH;
    const unsigned short* Kbh = KT + (size_t)bh * NPIX * DH;
    const unsigned short* Vbh = Vd + (size_t)bh * DH * NPIX;
    const unsigned short* Ph  = PT + (size_t)h * NPIX * DH;

    const int tid  = (int)threadIdx.x;
    const int wave = tid >> 5;
    const int lane = tid & 31;
    const int l16 = lane & 15, lhalf = lane >> 4;

    // ---- Phase 1: wave w owns column strip [w*128, w*128+128) of S
    v16bf A1[2][2], A2[2][2];   // A-frags for 2 row-tiles x 2 k-steps (d=64)
    #pragma unroll
    for (int rt = 0; rt < 2; ++rt) {
        #pragma unroll
        for (int ks = 0; ks < 2; ++ks) {
            const int row = n0 + rt * 16 + l16;
            const int k0  = ks * 32 + lhalf * 8;
            A1[rt][ks] = ldfragA(Qbh + (size_t)row * DH + k0);  // Q^T rows (cc term)
            A2[rt][ks] = ldfragA(Ph  + (size_t)row * DH + k0);  // pos^T rows (cp term)
        }
    }
    #pragma unroll
    for (int ct = 0; ct < 8; ++ct) {
        const int m0 = wave * 128 + ct * 16;
        v8f acc0 = { 0.f, 0.f, 0.f, 0.f, 0.f, 0.f, 0.f, 0.f };
        v8f acc1 = acc0;
        #pragma unroll
        for (int ks = 0; ks < 2; ++ks) {
            const int col = m0 + l16;
            const int kb  = ks * 32 + lhalf * 8;
            v16bf B1 = ldfragB(Kbh + (size_t)col * DH + kb);    // K columns
            v16bf B2 = ldfragB(Qbh + (size_t)col * DH + kb);    // Q columns
            acc0 = bfwmma(A1[0][ks], B1, acc0);
            acc0 = bfwmma(A2[0][ks], B2, acc0);
            acc1 = bfwmma(A1[1][ks], B1, acc1);
            acc1 = bfwmma(A2[1][ks], B2, acc1);
        }
        #pragma unroll
        for (int r = 0; r < 8; ++r) {
            S[(r + lhalf * 8) * S_PITCH + m0 + l16]      = acc0[r];
            S[(16 + r + lhalf * 8) * S_PITCH + m0 + l16] = acc1[r];
        }
    }
    __syncthreads();

    // ---- Phase 2: softmax over each of the 32 rows; 8 lanes per row
    {
        const int row = tid >> 3;     // 0..31
        const int seg = tid & 7;      // column class (mod 8)
        float* Srow = S + row * S_PITCH;
        float mx = -3.0e38f;
        #pragma unroll 4
        for (int j = 0; j < 128; ++j) mx = fmaxf(mx, Srow[seg + j * 8]);
        #pragma unroll
        for (int d = 1; d < 8; d <<= 1) mx = fmaxf(mx, __shfl_xor(mx, d, 32));
        float sum = 0.f;
        #pragma unroll 4
        for (int j = 0; j < 128; ++j) {
            float e = __expf(Srow[seg + j * 8] - mx);
            Srow[seg + j * 8] = e;
            sum += e;
        }
        #pragma unroll
        for (int d = 1; d < 8; d <<= 1) sum += __shfl_xor(sum, d, 32);
        const float inv = 1.f / sum;
        unsigned short* Prow = Pr + row * PR_PITCH;
        #pragma unroll 4
        for (int j = 0; j < 128; ++j)
            Prow[seg + j * 8] = f2bf(Srow[seg + j * 8] * inv);
    }
    __syncthreads();   // all waves done reading S -> its LDS space is reusable

    // ---- Phase V: async-copy V[64 rows][2048 B] into LDS (padded pitch)
    // 8192 16-byte chunks; thread t moves chunks t, t+256, ... (32 per thread);
    // each wave-level instruction moves 32 lanes x 16 B = 512 B asynchronously.
    {
        #pragma unroll
        for (int i = 0; i < 32; ++i) {
            const int chunk = tid + i * 256;
            const int r = chunk >> 7;          // V row (d index)
            const int j = chunk & 127;         // 16B chunk within row
            const unsigned long long ga =
                (unsigned long long)((const char*)Vbh + (size_t)r * 2048 + j * 16);
            auto lp = (__attribute__((address_space(3))) char*)
                          (smem + (size_t)r * (V_PITCH * 2) + j * 16);
            asm volatile("global_load_async_to_lds_b128 %0, %1, off"
                         :: "v"(lp), "v"(ga) : "memory");
        }
        asm volatile("s_wait_asynccnt 0x0" ::: "memory");
    }
    __syncthreads();

    // ---- Phase 3: O tile per wave (2 row-tiles x 4 d-tiles = 8 waves), all LDS
    {
        const unsigned short* Vlds = (const unsigned short*)smem;  // [64][V_PITCH]
        const int rt = wave & 1;
        const int dt = wave >> 1;
        v8f acc = { 0.f, 0.f, 0.f, 0.f, 0.f, 0.f, 0.f, 0.f };
        const unsigned short* Vrow = Vlds + (size_t)(dt * 16 + l16) * V_PITCH;
        const unsigned short* Arow = Pr + (rt * 16 + l16) * PR_PITCH;
        #pragma unroll 8
        for (int ks = 0; ks < 32; ++ks) {
            v16bf A = ldfragA(Arow + ks * 32 + lhalf * 8);   // probs (LDS)
            v16bf B = ldfragB(Vrow + ks * 32 + lhalf * 8);  // V (LDS)
            acc = bfwmma(A, B, acc);
        }
        const int c  = h * DH + dt * 16 + l16;
        const int nb = n0 + rt * 16 + lhalf * 8;
        const size_t off = ((size_t)b * C_DIM + c) * NPIX + nb;
        #pragma unroll
        for (int r = 0; r < 8; ++r)
            out[off + r] = acc[r] + x[off + r];   // residual
    }
}

// ---------------------------------------------------------------------------
extern "C" void kernel_launch(void* const* d_in, const int* in_sizes, int n_in,
                              void* d_out, int out_size, void* d_ws, size_t ws_size,
                              hipStream_t stream) {
    (void)in_sizes; (void)n_in; (void)out_size; (void)ws_size;
    const float* x     = (const float*)d_in[0];
    const float* Wq    = (const float*)d_in[1];
    const float* bq    = (const float*)d_in[2];
    const float* Wk    = (const float*)d_in[3];
    const float* bk    = (const float*)d_in[4];
    const float* Wv    = (const float*)d_in[5];
    const float* bv    = (const float*)d_in[6];
    const float* rel_h = (const float*)d_in[7];
    const float* rel_w = (const float*)d_in[8];
    // d_in[9] (reg_qk), d_in[10] (reg_v): dropped by out[:, :head] -> unused.

    // Workspace carve-up (~69.7 MB total, 256B aligned regions)
    size_t off = 0;
    auto carve = [&](size_t bytes) -> char* {
        char* p = (char*)d_ws + off;
        off += (bytes + 255) & ~(size_t)255;
        return p;
    };
    unsigned short* xT      = (unsigned short*)carve((size_t)BATCH * NPIX * C_DIM * 2);
    unsigned short* Wcat    = (unsigned short*)carve((size_t)OCH * C_DIM * 2);
    float*          biascat = (float*)carve((size_t)OCH * 4);
    unsigned short* QT      = (unsigned short*)carve((size_t)BATCH * NHEAD * NPIX * DH * 2);
    unsigned short* KT      = (unsigned short*)carve((size_t)BATCH * NHEAD * NPIX * DH * 2);
    unsigned short* Vd      = (unsigned short*)carve((size_t)BATCH * NHEAD * DH * NPIX * 2);
    unsigned short* PT      = (unsigned short*)carve((size_t)NHEAD * NPIX * DH * 2);

    prep_x_kernel<<<(BATCH * NPIX * C_DIM) / 256, 256, 0, stream>>>(x, xT);
    prep_w_kernel<<<(OCH * C_DIM) / 256, 256, 0, stream>>>(Wq, Wk, Wv, bq, bk, bv,
                                                           Wcat, biascat);
    prep_p_kernel<<<(NHEAD * NPIX * DH) / 256, 256, 0, stream>>>(rel_h, rel_w, PT);

    // 16x16 tiles: 16 batches x 96 o-tiles x 64 n-tiles = 98304 waves / 8 = 12288 blocks
    qkv_gemm_kernel<<<12288, 256, 0, stream>>>(xT, Wcat, biascat, QT, KT, Vd);

    // 16 b x 8 h x 32 row-slabs = 4096 blocks; ~194.5 KB dynamic LDS
    attn_kernel<<<4096, 256, SMEM_BYTES, stream>>>(QT, KT, Vd, PT, x, (float*)d_out);
}
